// LinearAttention_25872882991986
// MI455X (gfx1250) — matrix operand order repaired and checked by reading, not compiled, not run on hardware
//
#include <hip/hip_runtime.h>
#include <hip/hip_bf16.h>

typedef __attribute__((ext_vector_type(16))) _Float16 v16h;
typedef __attribute__((ext_vector_type(8)))  _Float16 v8h;
typedef __attribute__((ext_vector_type(8)))  float    v8f;

// ---------------- workspace layout (bytes) ----------------
#define WQKV_H_OFF   0u          // 768*256 f16      = 393216
#define WOUT_H_OFF   393216u     // 256*256 f16      = 131072
#define CTXN_OFF     524288u     // 8*8*32*32 f16    = 131072
#define PKSUM_OFF    655360u     // 8*64*256 f32     = 524288
#define PCTX_OFF     1179648u    // 8*64*8*1024 f32  = 16777216
#define QSOFT_OFF    17956864u   // 8*8*16384*32 f16 = 67108864
// total ~81.1 MB

// ---------------- WMMA fragment loaders ----------------
// A operand (16x32, f16): memory is row-major [m][k], stride in halfs (mult of 8).
// element e: value = M[lo][16*(e/8) + 8*hi + (e%8)]  -> two 16B loads.
__device__ __forceinline__ v16h load_a_frag(const _Float16* p, int stride) {
  const int lane = threadIdx.x & 31;
  const int lo = lane & 15;
  const int hi = (lane >> 4) & 1;
  const _Float16* r0 = p + lo * stride + 8 * hi;
  v8h x0 = *(const v8h*)(r0);
  v8h x1 = *(const v8h*)(r0 + 16);
  v16h r;
#pragma unroll
  for (int i = 0; i < 8; ++i) { r[i] = x0[i]; r[i + 8] = x1[i]; }
  return r;
}

// B operand (32x16, f16): memory holds B^T row-major [n][k], stride in halfs.
// element e: value = B[16*(e/8) + 8*hi + (e%8)][lo]  -> two 16B loads.
__device__ __forceinline__ v16h load_b_frag(const _Float16* p, int stride) {
  const int lane = threadIdx.x & 31;
  const int lo = lane & 15;
  const int hi = (lane >> 4) & 1;
  const _Float16* r0 = p + lo * stride + 8 * hi;
  v8h x0 = *(const v8h*)(r0);
  v8h x1 = *(const v8h*)(r0 + 16);
  v16h r;
#pragma unroll
  for (int i = 0; i < 8; ++i) { r[i] = x0[i]; r[i + 8] = x1[i]; }
  return r;
}

__device__ __forceinline__ v8f wmma_f16(v16h a, v16h b, v8f c) {
  return __builtin_amdgcn_wmma_f32_16x16x32_f16(false, a, false, b, (short)0, c,
                                                false, false);
}

// ---------------- k0: weights fp32 -> f16 ----------------
__global__ __launch_bounds__(256) void k0_prep(const float* __restrict__ wqkv,
                                               const float* __restrict__ wout,
                                               _Float16* __restrict__ wqkv_h,
                                               _Float16* __restrict__ wout_h) {
  int i = blockIdx.x * 256 + threadIdx.x;
  if (i < 768 * 256) wqkv_h[i] = (_Float16)wqkv[i];
  if (i < 256 * 256) wout_h[i] = (_Float16)wout[i];
}

// ---------------- k1a: q GEMM + channel softmax ----------------
// grid: 8 batches * 512 tiles (32 positions each)
__global__ __launch_bounds__(256) void k1a_q(const float* __restrict__ x,
                                             const _Float16* __restrict__ wqkv_h,
                                             const float* __restrict__ bqkv,
                                             _Float16* __restrict__ qsoft) {
  __shared__ __align__(16) _Float16 xs[32 * 264];   // [n][c], padded stride
  __shared__ __align__(16) _Float16 qs[256 * 40];   // [m][n], padded stride
  const int tid = threadIdx.x;
  const int b = blockIdx.x >> 9;
  const int tile = blockIdx.x & 511;
  const int n0 = tile * 32;
  const int w = tid >> 5, lane = tid & 31, lo = lane & 15, hi = (lane >> 4) & 1;

  // stage x tile (fp32 -> f16, transpose to [n][c])
  {
    const float* xb = x + (size_t)b * 256 * 16384 + n0;
    for (int i = tid; i < 2048; i += 256) {
      int c = i >> 3, n4 = (i & 7) * 4;
      float4 v = *(const float4*)(xb + (size_t)c * 16384 + n4);
      xs[(n4 + 0) * 264 + c] = (_Float16)v.x;
      xs[(n4 + 1) * 264 + c] = (_Float16)v.y;
      xs[(n4 + 2) * 264 + c] = (_Float16)v.z;
      xs[(n4 + 3) * 264 + c] = (_Float16)v.w;
    }
  }
  __syncthreads();

  // GEMM: q rows 0..255, wave w -> M-tiles {w, w+8}
  v8f acc[2][2] = {};
  for (int kk = 0; kk < 8; ++kk) {
    v16h b0 = load_b_frag(xs + kk * 32, 264);
    v16h b1 = load_b_frag(xs + 16 * 264 + kk * 32, 264);
    v16h a0 = load_a_frag(wqkv_h + (size_t)(16 * w) * 256 + kk * 32, 256);
    v16h a1 = load_a_frag(wqkv_h + (size_t)(16 * (w + 8)) * 256 + kk * 32, 256);
    acc[0][0] = wmma_f16(a0, b0, acc[0][0]);
    acc[0][1] = wmma_f16(a0, b1, acc[0][1]);
    acc[1][0] = wmma_f16(a1, b0, acc[1][0]);
    acc[1][1] = wmma_f16(a1, b1, acc[1][1]);
  }
#pragma unroll
  for (int mi = 0; mi < 2; ++mi) {
    int mbase = 16 * w + 128 * mi;
#pragma unroll
    for (int j = 0; j < 8; ++j) {
      float bb = bqkv[mbase + j + 8 * hi];
      qs[(mbase + j + 8 * hi) * 40 + lo]      = (_Float16)(acc[mi][0][j] + bb);
      qs[(mbase + j + 8 * hi) * 40 + 16 + lo] = (_Float16)(acc[mi][1][j] + bb);
    }
  }
  __syncthreads();

  // per-(head,pos) softmax over 32 channels; thread = (h, n)
  {
    int h = tid >> 5, n = tid & 31;
    float vals[32];
    float mx = -1e30f;
#pragma unroll
    for (int c = 0; c < 32; ++c) {
      vals[c] = (float)qs[(32 * h + c) * 40 + n];
      mx = fmaxf(mx, vals[c]);
    }
    float s = 0.f;
#pragma unroll
    for (int c = 0; c < 32; ++c) { vals[c] = __expf(vals[c] - mx); s += vals[c]; }
    float inv = 1.f / s;
    _Float16* dst = qsoft + ((size_t)(b * 8 + h) * 16384 + n0 + n) * 32;
#pragma unroll
    for (int q4 = 0; q4 < 4; ++q4) {
      v8h t;
#pragma unroll
      for (int i = 0; i < 8; ++i) t[i] = (_Float16)(vals[8 * q4 + i] * inv);
      *(v8h*)(dst + 8 * q4) = t;
    }
  }
}

// ---------------- k1b: k/v GEMM + exp + context partials ----------------
// grid: 8 batches * 64 chunks (256 positions each, 8 subtiles of 32)
__global__ __launch_bounds__(256) void k1b_kv(const float* __restrict__ x,
                                              const _Float16* __restrict__ wqkv_h,
                                              const float* __restrict__ bqkv,
                                              float* __restrict__ pksum,
                                              float* __restrict__ pctx) {
  __shared__ __align__(16) _Float16 xs[32 * 264];
  __shared__ __align__(16) _Float16 qs[256 * 40];  // one 4-head group: k(128)+v(128)
  const int tid = threadIdx.x;
  const int b = blockIdx.x >> 6;
  const int chunk = blockIdx.x & 63;
  const int w = tid >> 5, lane = tid & 31, lo = lane & 15, hi = (lane >> 4) & 1;
  const int hl = w >> 1, mi = w & 1;

  v8f ctxacc[2][2] = {};          // [group][d-tile]
  float ksacc[2] = {0.f, 0.f};

  int rowbase[4];
#pragma unroll
  for (int sl = 0; sl < 4; ++sl) {
    int g = sl >> 1;
    int sr = 16 * (w + 8 * (sl & 1));               // stage row base
    rowbase[sl] = (sr < 128) ? (256 + 128 * g + sr) // k rows
                             : (384 + 128 * g + sr);// v rows (512 + sr-128)
  }

  for (int s = 0; s < 8; ++s) {
    const int n0 = chunk * 256 + s * 32;
    {
      const float* xb = x + (size_t)b * 256 * 16384 + n0;
      for (int i = tid; i < 2048; i += 256) {
        int c = i >> 3, n4 = (i & 7) * 4;
        float4 v = *(const float4*)(xb + (size_t)c * 16384 + n4);
        xs[(n4 + 0) * 264 + c] = (_Float16)v.x;
        xs[(n4 + 1) * 264 + c] = (_Float16)v.y;
        xs[(n4 + 2) * 264 + c] = (_Float16)v.z;
        xs[(n4 + 3) * 264 + c] = (_Float16)v.w;
      }
    }
    __syncthreads();

    // GEMM all 4 slots (both groups), 512 k/v rows total across WG
    v8f acc[4][2] = {};
    for (int kk = 0; kk < 8; ++kk) {
      v16h b0 = load_b_frag(xs + kk * 32, 264);
      v16h b1 = load_b_frag(xs + 16 * 264 + kk * 32, 264);
#pragma unroll
      for (int sl = 0; sl < 4; ++sl) {
        v16h a = load_a_frag(wqkv_h + (size_t)rowbase[sl] * 256 + kk * 32, 256);
        acc[sl][0] = wmma_f16(a, b0, acc[sl][0]);
        acc[sl][1] = wmma_f16(a, b1, acc[sl][1]);
      }
    }

#pragma unroll
    for (int g = 0; g < 2; ++g) {
      // spill group g (k rows -> qs[0..127], v rows -> qs[128..255])
#pragma unroll
      for (int which = 0; which < 2; ++which) {
        int sl = 2 * g + which;
        int srbase = 16 * (w + 8 * which);
#pragma unroll
        for (int j = 0; j < 8; ++j) {
          float bb = bqkv[rowbase[sl] + j + 8 * hi];
          qs[(srbase + j + 8 * hi) * 40 + lo]      = (_Float16)(acc[sl][0][j] + bb);
          qs[(srbase + j + 8 * hi) * 40 + 16 + lo] = (_Float16)(acc[sl][1][j] + bb);
        }
      }
      __syncthreads();

      // exp(k) in place + row-sum partials (rows 0..127 = k of this group)
      if (tid < 128) {
        float ks = 0.f;
#pragma unroll
        for (int n = 0; n < 32; ++n) {
          float v = __expf((float)qs[tid * 40 + n]);
          _Float16 vh = (_Float16)v;
          qs[tid * 40 + n] = vh;
          ks += (float)vh;   // keep ksum consistent with f16-rounded E
        }
        ksacc[g] += ks;
      }
      __syncthreads();

      // context partial: ctx[c][d] += E(32xn) * V(32xn)^T via WMMA
      {
        v16h a  = load_a_frag(qs + (32 * hl + 16 * mi) * 40, 40);
        v16h b0 = load_b_frag(qs + (128 + 32 * hl) * 40, 40);
        v16h b1 = load_b_frag(qs + (128 + 32 * hl + 16) * 40, 40);
        ctxacc[g][0] = wmma_f16(a, b0, ctxacc[g][0]);
        ctxacc[g][1] = wmma_f16(a, b1, ctxacc[g][1]);
      }
      __syncthreads();
    }
  }

  // write partials
  if (tid < 128) {
    float* pk = pksum + (size_t)(b * 64 + chunk) * 256;
    pk[tid]       = ksacc[0];
    pk[128 + tid] = ksacc[1];
  }
#pragma unroll
  for (int g = 0; g < 2; ++g) {
    int head = 4 * g + hl;
    float* pc = pctx + ((size_t)(b * 64 + chunk) * 8 + head) * 1024;
#pragma unroll
    for (int ni = 0; ni < 2; ++ni)
#pragma unroll
      for (int j = 0; j < 8; ++j)
        pc[(16 * mi + j + 8 * hi) * 32 + 16 * ni + lo] = ctxacc[g][ni][j];
  }
}

// ---------------- k2: reduce partials, normalize, transpose to f16 ----------------
// grid: 64 = (b,h)
__global__ __launch_bounds__(256) void k2_reduce(const float* __restrict__ pksum,
                                                 const float* __restrict__ pctx,
                                                 _Float16* __restrict__ ctxn) {
  __shared__ float ksh[32];
  const int tid = threadIdx.x;
  const int b = blockIdx.x >> 3, h = blockIdx.x & 7;
  const int g = h >> 2, hl = h & 3;
  if (tid < 32) {
    float s = 0.f;
    for (int ch = 0; ch < 64; ++ch)
      s += pksum[(size_t)(b * 64 + ch) * 256 + 128 * g + 32 * hl + tid];
    ksh[tid] = s;
  }
  __syncthreads();
  for (int e = tid; e < 1024; e += 256) {
    int c = e >> 5, d = e & 31;
    float s = 0.f;
    for (int ch = 0; ch < 64; ++ch)
      s += pctx[((size_t)(b * 64 + ch) * 8 + h) * 1024 + e];
    // store transposed [d][c] so it is a ready A-operand for k3
    ctxn[(size_t)(b * 8 + h) * 1024 + d * 32 + c] = (_Float16)(s / ksh[c]);
  }
}

// ---------------- k3: attn = ctx^T * q_soft, then w_out projection ----------------
// grid: 8 batches * 256 tiles (64 positions each)
__global__ __launch_bounds__(256) void k3_out(const _Float16* __restrict__ qsoft,
                                              const _Float16* __restrict__ ctxn,
                                              const _Float16* __restrict__ wout_h,
                                              const float* __restrict__ bout,
                                              float* __restrict__ out) {
  __shared__ __align__(16) _Float16 attn_s[64 * 264];  // [n][c]
  const int tid = threadIdx.x;
  const int b = blockIdx.x >> 8;
  const int tile = blockIdx.x & 255;
  const int n0 = tile * 64;
  const int w = tid >> 5, lane = tid & 31, lo = lane & 15, hi = (lane >> 4) & 1;

  // phase A: per-head attn tile; wave -> (n-tile = w&3, head-half = w>>2)
  {
    int nt = w & 3;
    int hh = w >> 2;
#pragma unroll
    for (int q = 0; q < 4; ++q) {
      int h = 4 * hh + q;
      v16h bf = load_b_frag(
          qsoft + ((size_t)(b * 8 + h) * 16384 + n0 + 16 * nt) * 32, 32);
#pragma unroll
      for (int mi2 = 0; mi2 < 2; ++mi2) {
        v16h af = load_a_frag(ctxn + (size_t)(b * 8 + h) * 1024 + 16 * mi2 * 32, 32);
        v8f d = {};
        d = wmma_f16(af, bf, d);
        v8h t;
#pragma unroll
        for (int j = 0; j < 8; ++j) t[j] = (_Float16)d[j];
        // transpose-store: row n = 16*nt+lo, cols 32h+16*mi2+8*hi .. +7 (contiguous)
        *(v8h*)(attn_s + (16 * nt + lo) * 264 + 32 * h + 16 * mi2 + 8 * hi) = t;
      }
    }
  }
  __syncthreads();

  // phase B: out = w_out(256x256) * attn(256x64) + b_out
#pragma unroll
  for (int ms = 0; ms < 2; ++ms) {
    int mt = w + 8 * ms;
    v8f acc[4] = {};
    for (int kk = 0; kk < 8; ++kk) {
      v16h a = load_a_frag(wout_h + (size_t)(16 * mt) * 256 + kk * 32, 256);
#pragma unroll
      for (int nt = 0; nt < 4; ++nt) {
        v16h bf = load_b_frag(attn_s + 16 * nt * 264 + kk * 32, 264);
        acc[nt] = wmma_f16(a, bf, acc[nt]);
      }
    }
#pragma unroll
    for (int nt = 0; nt < 4; ++nt)
#pragma unroll
      for (int j = 0; j < 8; ++j) {
        int m = 16 * mt + j + 8 * hi;
        out[((size_t)b * 256 + m) * 16384 + n0 + 16 * nt + lo] = acc[nt][j] + bout[m];
      }
  }
}

// ---------------- launch ----------------
extern "C" void kernel_launch(void* const* d_in, const int* in_sizes, int n_in,
                              void* d_out, int out_size, void* d_ws, size_t ws_size,
                              hipStream_t stream) {
  const float* x    = (const float*)d_in[0];
  const float* wqkv = (const float*)d_in[1];
  const float* bqkv = (const float*)d_in[2];
  const float* wout = (const float*)d_in[3];
  const float* bout = (const float*)d_in[4];
  char* ws = (char*)d_ws;
  _Float16* wqkv_h = (_Float16*)(ws + WQKV_H_OFF);
  _Float16* wout_h = (_Float16*)(ws + WOUT_H_OFF);
  _Float16* ctxn   = (_Float16*)(ws + CTXN_OFF);
  float*    pksum  = (float*)(ws + PKSUM_OFF);
  float*    pctx   = (float*)(ws + PCTX_OFF);
  _Float16* qsoft  = (_Float16*)(ws + QSOFT_OFF);
  float* out = (float*)d_out;

  k0_prep<<<768, 256, 0, stream>>>(wqkv, wout, wqkv_h, wout_h);
  k1a_q<<<4096, 256, 0, stream>>>(x, wqkv_h, bqkv, qsoft);
  k1b_kv<<<512, 256, 0, stream>>>(x, wqkv_h, bqkv, pksum, pctx);
  k2_reduce<<<64, 256, 0, stream>>>(pksum, pctx, ctxn);
  k3_out<<<2048, 256, 0, stream>>>(qsoft, ctxn, wout_h, bout, out);
}
